// Head_54305566491157
// MI455X (gfx1250) — matrix operand order repaired and checked by reading, compiled, not run on hardware
//
#include <hip/hip_runtime.h>

// ---------------------------------------------------------------------------
// Causal single-head attention, CDNA5 (gfx1250), wave32 WMMA f16.
// B=512, T=256, C=384, H=64.  SCALE = 1/8.
// ---------------------------------------------------------------------------

typedef _Float16 v16h __attribute__((ext_vector_type(16)));
typedef _Float16 v8h  __attribute__((ext_vector_type(8)));
typedef float    v8f  __attribute__((ext_vector_type(8)));
typedef float    v4f  __attribute__((ext_vector_type(4)));

#define WMMA16(a, b, c)                                                     \
  __builtin_amdgcn_wmma_f32_16x16x32_f16(false, (a), false, (b), (short)0,  \
                                         (c), false, false)

namespace {
constexpr int Bn = 512;
constexpr int Tn = 256;
constexpr int Cn = 384;
constexpr int Hn = 64;
constexpr float SCALE = 0.125f;           // 1/sqrt(64)
constexpr float L2E   = 1.44269504088896340736f;
}  // namespace

__device__ __forceinline__ v16h load16h(const _Float16* p) {
  v8h lo = *(const v8h*)(p);
  v8h hi = *(const v8h*)(p + 8);
  return __builtin_shufflevector(lo, hi, 0, 1, 2, 3, 4, 5, 6, 7, 8, 9, 10, 11,
                                 12, 13, 14, 15);
}

// ---------------------------------------------------------------------------
// Kernel 1: repack Wq/Wk/Wv [C,H] fp32 into WMMA B-matrix per-lane layout, f16.
// Tile (m, nt, kt): lane L owns column n = nt*16 + L%16, K rows
// kt*32 + (L/16)*8 + (j&7) + (j>>3)*16, j=0..15 stored contiguously (32B per lane).
// ---------------------------------------------------------------------------
__global__ void __launch_bounds__(32)
pack_weights(const float* __restrict__ Wq, const float* __restrict__ Wk,
             const float* __restrict__ Wv, _Float16* __restrict__ wpack) {
  const int blk = blockIdx.x;        // 0..143  (3 * 4 * 12)
  const int m   = blk / 48;          // matrix: 0=q,1=k,2=v
  const int nt  = (blk % 48) / 12;   // H tile
  const int kt  = blk % 12;          // C tile of 32
  const float* W = (m == 0) ? Wq : (m == 1) ? Wk : Wv;

  const int lane = threadIdx.x;
  const int lr = lane & 15, half = lane >> 4;
  const int n = nt * 16 + lr;
  const int kbase = kt * 32 + half * 8;

  _Float16* dst = wpack + (size_t)(((m * 4 + nt) * 12 + kt) * 512 + lane * 16);
#pragma unroll
  for (int j = 0; j < 16; ++j)
    dst[j] = (_Float16)W[(size_t)(kbase + (j & 7) + (j >> 3) * 16) * Hn + n];
}

// ---------------------------------------------------------------------------
// Kernel 2: q/k/v projections.  One wave per (b, 16-row tile).
// All 12 B tiles of a K-step are loaded into distinct registers first so the
// compiler can clause the loads and drain them with incremental waits while
// the 12 WMMAs issue.  144 WMMAs per wave.
// Outputs: qh, kh row-major [B,T,H] f16; v transposed vth [B,H,T] f16.
// ---------------------------------------------------------------------------
__global__ void __launch_bounds__(32)
qkv_proj(const float* __restrict__ x, const _Float16* __restrict__ wpack,
         _Float16* __restrict__ qh, _Float16* __restrict__ kh,
         _Float16* __restrict__ vth) {
  const int blk = blockIdx.x;
  const int b = blk >> 4;
  const int t0 = (blk & 15) * 16;
  const int lane = threadIdx.x;
  const int lr = lane & 15, half = lane >> 4;

  const float* xrow = x + (size_t)(b * Tn + t0 + lr) * Cn;

  v8f acc[3][4] = {};

  for (int kt = 0; kt < 12; ++kt) {
    // A tile: lane holds row t0+lr, K chunks [half*8, half*8+8) and +16.
    const float* xp = xrow + kt * 32 + half * 8;
    v4f f0 = *(const v4f*)(xp);
    v4f f1 = *(const v4f*)(xp + 4);
    v4f f2 = *(const v4f*)(xp + 16);
    v4f f3 = *(const v4f*)(xp + 20);

    // Stage all 12 B tiles (q/k/v x 4 H-tiles) into distinct registers.
    const _Float16* wbase = wpack + (size_t)kt * 512 + lane * 16;
    v16h bm[12];
#pragma unroll
    for (int i = 0; i < 12; ++i)
      bm[i] = load16h(wbase + (size_t)(i * 12) * 512);

    if (kt < 11) __builtin_prefetch(xp + 32, 0, 0);

    v16h a;
#pragma unroll
    for (int i = 0; i < 4; ++i) {
      a[i]      = (_Float16)f0[i];
      a[4 + i]  = (_Float16)f1[i];
      a[8 + i]  = (_Float16)f2[i];
      a[12 + i] = (_Float16)f3[i];
    }

#pragma unroll
    for (int i = 0; i < 12; ++i)
      acc[i >> 2][i & 3] = WMMA16(a, bm[i], acc[i >> 2][i & 3]);
  }

  // C layout: lane<16 -> M=r, N=lr ; lane>=16 -> M=8+r, N=lr.
#pragma unroll
  for (int nt = 0; nt < 4; ++nt) {
    const int col = nt * 16 + lr;
#pragma unroll
    for (int r = 0; r < 8; ++r) {
      const int row = t0 + half * 8 + r;
      qh[(size_t)(b * Tn + row) * Hn + col] = (_Float16)acc[0][nt][r];
      kh[(size_t)(b * Tn + row) * Hn + col] = (_Float16)acc[1][nt][r];
      vth[((size_t)b * Hn + col) * Tn + row] = (_Float16)acc[2][nt][r];
    }
  }
}

// ---------------------------------------------------------------------------
// Kernel 3: flash attention.  One wave per (b, 16-query-row tile).
// Per 32-key chunk: preload 4 K^T tiles AND 4 V tiles (V is independent of
// softmax, so its loads overlap the S WMMAs and softmax VALU work), then
// S = Q K^T (4 WMMA), base-2 online softmax, P via LDS (C->A layout),
// O += P V (4 WMMA).
// ---------------------------------------------------------------------------
__global__ void __launch_bounds__(32)
attn(const _Float16* __restrict__ qh, const _Float16* __restrict__ kh,
     const _Float16* __restrict__ vth, float* __restrict__ out) {
  __shared__ _Float16 plds[16 * 32];  // [qrow][key] row-major, 1 KB

  const int blk = blockIdx.x;
  const int b = blk >> 4;
  const int t0 = (blk & 15) * 16;
  const int lane = threadIdx.x;
  const int lr = lane & 15, half = lane >> 4;

  // Q A-tiles: ks=0 -> h in [0,32), ks=1 -> h in [32,64).
  v16h qa[2];
#pragma unroll
  for (int ks = 0; ks < 2; ++ks) {
    const _Float16* qp =
        qh + (size_t)(b * Tn + t0 + lr) * Hn + ks * 32 + half * 16;
    qa[ks] = __builtin_shufflevector(*(const v8h*)qp, *(const v8h*)(qp + 8),
                                     0, 1, 2, 3, 4, 5, 6, 7, 8, 9, 10, 11, 12,
                                     13, 14, 15);
  }

  v8f o[4] = {};
  float mrow[8], lsum[8];
#pragma unroll
  for (int r = 0; r < 8; ++r) {
    mrow[r] = -__builtin_inff();
    lsum[r] = 0.0f;
  }

  const float kS = SCALE * L2E;  // logits in log2 domain
  const int jmax = (t0 + 15) >> 5;

  for (int j = 0; j <= jmax; ++j) {
    const int key0 = j * 32;

    // ---- preload K^T tiles (kb[ks][jn]) and V tiles (vb[nt]) -------------
    v16h kb[2][2];
#pragma unroll
    for (int ks = 0; ks < 2; ++ks) {
#pragma unroll
      for (int jn = 0; jn < 2; ++jn)
        kb[ks][jn] = load16h(kh + (size_t)(b * Tn + key0 + jn * 16 + lr) * Hn +
                             ks * 32 + half * 16);
    }
    v16h vb[4];
#pragma unroll
    for (int nt = 0; nt < 4; ++nt)
      vb[nt] = load16h(vth + ((size_t)b * Hn + nt * 16 + lr) * Tn + key0 +
                       half * 16);

    if (j < jmax)
      __builtin_prefetch(kh + (size_t)(b * Tn + key0 + 32 + lr) * Hn, 0, 0);

    // ---- S = Q K^T -------------------------------------------------------
    v8f s0 = {}, s1 = {};
#pragma unroll
    for (int ks = 0; ks < 2; ++ks) {
      s0 = WMMA16(qa[ks], kb[ks][0], s0);
      s1 = WMMA16(qa[ks], kb[ks][1], s1);
    }

    // ---- causal mask (diagonal chunks only) ------------------------------
    if (key0 + 31 > t0) {
#pragma unroll
      for (int r = 0; r < 8; ++r) {
        const int row = t0 + half * 8 + r;
        if (key0 + lr > row)      s0[r] = -__builtin_inff();
        if (key0 + 16 + lr > row) s1[r] = -__builtin_inff();
      }
    }

    // ---- online softmax (rows live in 16-lane halves) --------------------
    float alpha[8];
#pragma unroll
    for (int r = 0; r < 8; ++r) {
      float rm = fmaxf(s0[r], s1[r]);
      rm = fmaxf(rm, __shfl_xor(rm, 1, 32));
      rm = fmaxf(rm, __shfl_xor(rm, 2, 32));
      rm = fmaxf(rm, __shfl_xor(rm, 4, 32));
      rm = fmaxf(rm, __shfl_xor(rm, 8, 32));
      const float mN = fmaxf(mrow[r], rm * kS);
      const float a_ = exp2f(mrow[r] - mN);
      const float e0 = exp2f(s0[r] * kS - mN);
      const float e1 = exp2f(s1[r] * kS - mN);
      plds[(half * 8 + r) * 32 + lr]      = (_Float16)e0;
      plds[(half * 8 + r) * 32 + 16 + lr] = (_Float16)e1;
      float ps = e0 + e1;
      ps += __shfl_xor(ps, 1, 32);
      ps += __shfl_xor(ps, 2, 32);
      ps += __shfl_xor(ps, 4, 32);
      ps += __shfl_xor(ps, 8, 32);
      lsum[r] = lsum[r] * a_ + ps;
      mrow[r] = mN;
      alpha[r] = a_;
    }
#pragma unroll
    for (int nt = 0; nt < 4; ++nt)
#pragma unroll
      for (int r = 0; r < 8; ++r) o[nt][r] *= alpha[r];

    __syncthreads();

    // ---- P (A-layout from LDS) and O += P V ------------------------------
    const _Float16* pl = &plds[lr * 32 + half * 16];
    v16h pa = __builtin_shufflevector(*(const v8h*)pl, *(const v8h*)(pl + 8),
                                      0, 1, 2, 3, 4, 5, 6, 7, 8, 9, 10, 11,
                                      12, 13, 14, 15);
#pragma unroll
    for (int nt = 0; nt < 4; ++nt) o[nt] = WMMA16(pa, vb[nt], o[nt]);

    __syncthreads();
  }

  // ---- epilogue: O / l ---------------------------------------------------
#pragma unroll
  for (int nt = 0; nt < 4; ++nt) {
    const int col = nt * 16 + lr;
#pragma unroll
    for (int r = 0; r < 8; ++r) {
      const int row = t0 + half * 8 + r;
      out[(size_t)(b * Tn + row) * Hn + col] = o[nt][r] / lsum[r];
    }
  }
}

// ---------------------------------------------------------------------------
// Launch: pack -> project -> attend (stream-ordered).
// Workspace: [wpack 144 KB][qh 16 MB][kh 16 MB][vth 16 MB]  (~50.6 MB).
// ---------------------------------------------------------------------------
extern "C" void kernel_launch(void* const* d_in, const int* in_sizes, int n_in,
                              void* d_out, int out_size, void* d_ws,
                              size_t ws_size, hipStream_t stream) {
  (void)in_sizes; (void)n_in; (void)out_size; (void)ws_size;
  const float* x  = (const float*)d_in[0];
  const float* Wq = (const float*)d_in[1];
  const float* Wk = (const float*)d_in[2];
  const float* Wv = (const float*)d_in[3];
  float* out = (float*)d_out;

  char* ws = (char*)d_ws;
  _Float16* wpack = (_Float16*)ws;                 // 3*4*12*512 f16 = 144 KB
  const size_t base = 256 * 1024;
  const size_t qkv = (size_t)Bn * Tn * Hn * sizeof(_Float16);
  _Float16* qh  = (_Float16*)(ws + base);
  _Float16* kh  = (_Float16*)(ws + base + qkv);
  _Float16* vth = (_Float16*)(ws + base + 2 * qkv);

  pack_weights<<<dim3(144), dim3(32), 0, stream>>>(Wq, Wk, Wv, wpack);
  qkv_proj<<<dim3(Bn * 16), dim3(32), 0, stream>>>(x, wpack, qh, kh, vth);
  attn<<<dim3(Bn * 16), dim3(32), 0, stream>>>(qh, kh, vth, out);
}
